// VAELinear_13211319403237
// MI455X (gfx1250) — matrix operand order repaired and checked by reading, not compiled, not run on hardware
//
#include <hip/hip_runtime.h>
#include <hip/hip_bf16.h>

#define IN_F   4096
#define OUT_F  4096
#define LATENT 16
#define HIDDEN 64
#define CB     16
#define TOKENS 128
#define NCHUNK (IN_F / 64)

typedef __attribute__((ext_vector_type(16))) __bf16 v16bf;
typedef __attribute__((ext_vector_type(8)))  __bf16 v8bf;
typedef __attribute__((ext_vector_type(8)))  float  v8f;

#define HAS_ASYNC_LDS __has_builtin(__builtin_amdgcn_global_load_async_to_lds_b128)
#define HAS_WAIT_ASYNC __has_builtin(__builtin_amdgcn_s_wait_asynccnt)

// builtin signature (from clang diagnostic): param1 = v4i addrspace(1)* (global),
// param2 = LDS side, then i32 offset, i32 cpol
typedef int v4i __attribute__((vector_size(16)));
typedef __attribute__((address_space(1))) v4i* gptr_v4i;
typedef __attribute__((address_space(3))) v4i* lptr_v4i;

static __device__ __forceinline__ void cp16_async(void* lds, const void* g) {
#if HAS_ASYNC_LDS
  __builtin_amdgcn_global_load_async_to_lds_b128(
      (gptr_v4i)g, (lptr_v4i)lds, /*offset=*/0, /*cpol=*/0);
#else
  *reinterpret_cast<float4*>(lds) = *reinterpret_cast<const float4*>(g);
#endif
}

static __device__ __forceinline__ void wait_async() {
#if HAS_ASYNC_LDS
# if HAS_WAIT_ASYNC
  __builtin_amdgcn_s_wait_asynccnt(0);
# else
  asm volatile("s_wait_asynccnt 0" ::: "memory");
# endif
#endif
}

static __device__ __forceinline__ v16bf cat8(v8bf lo, v8bf hi) {
  return __builtin_shufflevector(lo, hi, 0,1,2,3,4,5,6,7,8,9,10,11,12,13,14,15);
}
static __device__ __forceinline__ v16bf pack16(float4 a, float4 b, float4 c, float4 d) {
  v16bf r;
  r[0]=(__bf16)a.x;  r[1]=(__bf16)a.y;  r[2]=(__bf16)a.z;  r[3]=(__bf16)a.w;
  r[4]=(__bf16)b.x;  r[5]=(__bf16)b.y;  r[6]=(__bf16)b.z;  r[7]=(__bf16)b.w;
  r[8]=(__bf16)c.x;  r[9]=(__bf16)c.y;  r[10]=(__bf16)c.z; r[11]=(__bf16)c.w;
  r[12]=(__bf16)d.x; r[13]=(__bf16)d.y; r[14]=(__bf16)d.z; r[15]=(__bf16)d.w;
  return r;
}
static __device__ __forceinline__ v8f wmma_bf16(v16bf a, v16bf b, v8f c) {
  return __builtin_amdgcn_wmma_f32_16x16x32_bf16(false, a, false, b, (short)0, c, false, false);
}

__global__ __launch_bounds__(256)
void vae_linear_fused(const float* __restrict__ x,
                      const float* __restrict__ vq,
                      const float* __restrict__ w1,
                      const float* __restrict__ b1,
                      const float* __restrict__ w2,
                      const float* __restrict__ b2,
                      const float* __restrict__ bias,
                      float* __restrict__ out)
{
  // double-buffered fp32 staging (async DMA targets), bf16 intermediates
  __shared__ __align__(16) float  xsf[2][TOKENS * 64];   // 64 KB
  __shared__ __align__(16) float  vqf[2][64 * LATENT];   //  8 KB
  __shared__ __align__(16) __bf16 hbuf[64 * HIDDEN];     //  8 KB
  __shared__ __align__(16) __bf16 wtile[16 * 64];        //  2 KB

  const int tid     = threadIdx.x;
  const int lane    = tid & 31;
  const int wave    = tid >> 5;
  const int om_base = blockIdx.x * 4;          // 256 blocks cover om = 0..1023
  const bool hih    = (lane >= 16);
  const int  l15    = lane & 15;

  // ---- loop-invariant decoder B fragments ----
  const int mt1 = wave >> 1;        // layer-1 M tile (0..3)
  const int ntb = (wave & 1) * 2;   // layer-1 N tile base (0 or 2)

  v16bf bw1[2];
  float bias1[2];
  for (int j = 0; j < 2; ++j) {
    const int nt = ntb + j;
    v16bf f;
    if (!hih) {                      // lanes 0..15: K = 0..15 (real latent K)
      const int col = nt * 16 + lane;
      #pragma unroll
      for (int k = 0; k < 16; ++k) f[k] = (__bf16)w1[k * HIDDEN + col];
    } else {                         // lanes 16..31: K = 16..31 -> zero pad
      #pragma unroll
      for (int k = 0; k < 16; ++k) f[k] = (__bf16)0.0f;
    }
    bw1[j]  = f;
    bias1[j] = b1[nt * 16 + l15];
  }

  v16bf bw2[2];
  for (int ks = 0; ks < 2; ++ks) {
    v16bf f;
    const int kb = ks * 32 + (hih ? 16 : 0);
    #pragma unroll
    for (int k = 0; k < 16; ++k) f[k] = (__bf16)w2[(kb + k) * CB + l15];
    bw2[ks] = f;
  }
  const float bias2 = b2[l15];

  v8f acc;
  #pragma unroll
  for (int r = 0; r < 8; ++r) acc[r] = 0.0f;

  const int mt2 = wave;             // layer-2 M tile (waves 0..3)

  // ---- async staging of one 64-wide K chunk into buffer `buf` ----
  auto stage = [&](int chunk, int buf) {
    // vq tile first: this is the HBM-critical stream
    {
      const int b  = tid >> 2;                 // b = om_idx*16 + ib_idx
      const int l4 = (tid & 3) * 4;
      const int nb = (om_base + (b >> 4)) * 1024 + chunk * 16 + (b & 15);
      cp16_async(&vqf[buf][b * LATENT + l4], vq + (long)nb * LATENT + l4);
    }
    // x tile [128 x 64]
    {
      const int t  = tid >> 1;
      const int cb = (tid & 1) * 32;
      const float* src = x + t * IN_F + chunk * 64 + cb;
      float* dst = &xsf[buf][t * 64 + cb];
      #pragma unroll
      for (int j = 0; j < 8; ++j) cp16_async(dst + 4 * j, src + 4 * j);
    }
  };

  stage(0, 0);

  for (int chunk = 0; chunk < NCHUNK; ++chunk) {
    const int cur = chunk & 1;

    wait_async();        // our DMA into buffer `cur` complete
    __syncthreads();     // everyone's staging visible; prev-iter LDS reads done

    if (chunk + 1 < NCHUNK) stage(chunk + 1, cur ^ 1);   // prefetch next chunk

    // ---- decoder layer 1: h = relu(vq @ W1 + b1), K=16 padded to 32 ----
    {
      const float* vrow = &vqf[cur][(mt1 * 16 + l15) * LATENT];
      float4 z4 = make_float4(0.f, 0.f, 0.f, 0.f);
      float4 q0 = hih ? z4 : *reinterpret_cast<const float4*>(vrow);
      float4 q1 = hih ? z4 : *reinterpret_cast<const float4*>(vrow + 4);
      float4 q2 = hih ? z4 : *reinterpret_cast<const float4*>(vrow + 8);
      float4 q3 = hih ? z4 : *reinterpret_cast<const float4*>(vrow + 12);
      const v16bf a1 = pack16(q0, q1, q2, q3);   // K 16..31 zero
      v8f h0, h1;
      #pragma unroll
      for (int r = 0; r < 8; ++r) { h0[r] = 0.0f; h1[r] = 0.0f; }
      h0 = wmma_bf16(a1, bw1[0], h0);
      h1 = wmma_bf16(a1, bw1[1], h1);
      const int rb = mt1 * 16 + (hih ? 8 : 0);   // C rows: M = r (+8 hi lanes)
      #pragma unroll
      for (int r = 0; r < 8; ++r) {
        float v0 = h0[r] + bias1[0]; v0 = v0 > 0.0f ? v0 : 0.0f;
        float v1 = h1[r] + bias1[1]; v1 = v1 > 0.0f ? v1 : 0.0f;
        hbuf[(rb + r) * HIDDEN + (ntb + 0) * 16 + l15] = (__bf16)v0;
        hbuf[(rb + r) * HIDDEN + (ntb + 1) * 16 + l15] = (__bf16)v1;
      }
    }

    __syncthreads();

    // ---- decoder layer 2 + permute scatter into W tile (waves 0..3) ----
    if (wave < 4) {
      v8f c2;
      #pragma unroll
      for (int r = 0; r < 8; ++r) c2[r] = 0.0f;
      #pragma unroll
      for (int ks = 0; ks < 2; ++ks) {
        const int ko = ks * 32 + (hih ? 16 : 0);
        const __bf16* row = &hbuf[(mt2 * 16 + l15) * HIDDEN];
        v8bf lo = *reinterpret_cast<const v8bf*>(row + ko);
        v8bf hh = *reinterpret_cast<const v8bf*>(row + ko + 8);
        c2 = wmma_bf16(cat8(lo, hh), bw2[ks], c2);
      }
      // wb[b, c] -> Wt[o_local = 4*g + om_idx][i_local = 4*ib_idx + s]
      const int g = l15 >> 2, s = l15 & 3;
      const int rb = hih ? 8 : 0;
      #pragma unroll
      for (int r = 0; r < 8; ++r) {
        wtile[(g * 4 + mt2) * 64 + (rb + r) * 4 + s] = (__bf16)(c2[r] + bias2);
      }
    }

    __syncthreads();

    // ---- main GEMM: acc[16t x 16o] += x_tile @ Wt^T over this K chunk ----
    {
      const float* xrow = &xsf[cur][(wave * 16 + l15) * 64];
      const __bf16* wrow = &wtile[l15 * 64];
      #pragma unroll
      for (int ks = 0; ks < 2; ++ks) {
        const int ko = ks * 32 + (hih ? 8 : 0);
        float4 a0 = *reinterpret_cast<const float4*>(xrow + ko);
        float4 a1_ = *reinterpret_cast<const float4*>(xrow + ko + 4);
        float4 a2 = *reinterpret_cast<const float4*>(xrow + ko + 16);
        float4 a3 = *reinterpret_cast<const float4*>(xrow + ko + 20);
        const v16bf afrag = pack16(a0, a1_, a2, a3);
        const int bo = ks * 32 + (hih ? 8 : 0);
        v8bf blo = *reinterpret_cast<const v8bf*>(wrow + bo);
        v8bf bhi = *reinterpret_cast<const v8bf*>(wrow + bo + 16);
        acc = wmma_bf16(afrag, cat8(blo, bhi), acc);
      }
    }
  }

  // ---- epilogue: add output bias, store [128 x 16] tile ----
  {
    const int o  = om_base + (l15 & 3) + ((l15 >> 2) << 10);
    const float ob = bias[o];
    const int tb = wave * 16 + (hih ? 8 : 0);
    #pragma unroll
    for (int r = 0; r < 8; ++r) {
      out[(tb + r) * OUT_F + o] = acc[r] + ob;
    }
  }
}

extern "C" void kernel_launch(void* const* d_in, const int* in_sizes, int n_in,
                              void* d_out, int out_size, void* d_ws, size_t ws_size,
                              hipStream_t stream) {
  const float* x    = (const float*)d_in[0];
  const float* vq   = (const float*)d_in[1];
  const float* w1   = (const float*)d_in[2];
  const float* b1   = (const float*)d_in[3];
  const float* w2   = (const float*)d_in[4];
  const float* b2   = (const float*)d_in[5];
  const float* bias = (const float*)d_in[6];
  float* out = (float*)d_out;

  dim3 grid(256), block(256);
  vae_linear_fused<<<grid, block, 0, stream>>>(x, vq, w1, b1, w2, b2, bias, out);
}
